// HGCN_11587821765286
// MI455X (gfx1250) — matrix operand order repaired and checked by reading, not compiled, not run on hardware
//
#include <hip/hip_runtime.h>
#include <hip/hip_bf16.h>

#define DD 128
#define TM 32            // rows per hyplinear block
#define WSTR 136         // LDS stride for bf16 W (272B, 16B-aligned, bank-staggered)
#define XSTR 136         // LDS stride for bf16 x tile
#define MXSTR 129        // LDS stride for f32 mx tile (conflict-free row walks)
#define MIN_NORM 1e-15f
#define MAXNORM  0.996f              // 1 - 4e-3
#define ART_CLAMP (1.0f - 1e-7f)

typedef __bf16 bf16;
typedef __attribute__((ext_vector_type(16))) __bf16 v16bf;
typedef __attribute__((ext_vector_type(8)))  __bf16 v8bf;
typedef __attribute__((ext_vector_type(8)))  float  v8f;
typedef __attribute__((ext_vector_type(4)))  unsigned int v4u;
typedef __attribute__((ext_vector_type(8)))  int  v8i;
typedef __attribute__((ext_vector_type(4)))  int  v4i;

// ---------------------------------------------------------------------------
// encode: xh = proj(expmap0(x, c=1))   — one wave per row, shuffle reduction
// ---------------------------------------------------------------------------
__global__ __launch_bounds__(256)
void hgcn_encode(const float* __restrict__ in, float* __restrict__ out, int n) {
    int wid  = (blockIdx.x * blockDim.x + threadIdx.x) >> 5;
    int lane = threadIdx.x & 31;
    if (wid >= n) return;
    const float4 v = *(const float4*)&in[(size_t)wid * DD + lane * 4];
    float s = v.x*v.x + v.y*v.y + v.z*v.z + v.w*v.w;
    #pragma unroll
    for (int off = 16; off; off >>= 1) s += __shfl_xor(s, off, 32);
    float nn = fmaxf(sqrtf(s), MIN_NORM);
    float tn = tanhf(nn);
    float f  = tn / nn;                         // expmap0 factor (sc = 1)
    float pf = (tn > MAXNORM) ? MAXNORM / tn : 1.0f;   // proj
    float sc = f * pf;
    float4 o = {v.x*sc, v.y*sc, v.z*sc, v.w*sc};
    *(float4*)&out[(size_t)wid * DD + lane * 4] = o;
}

// ---------------------------------------------------------------------------
// W fp32 -> bf16 (once per layer) so the Tensor Data Mover can DMA it raw.
// ---------------------------------------------------------------------------
__global__ __launch_bounds__(256)
void hgcn_w2bf16(const float* __restrict__ w, bf16* __restrict__ wb) {
    int i = blockIdx.x * 256 + threadIdx.x;
    if (i < DD * DD) wb[i] = (bf16)w[i];
}

// ---------------------------------------------------------------------------
// hyplinear: fused mobius_matvec (WMMA bf16) + proj + mobius_add(hyp_b) +
//            proj + logmap0.  64 threads (2 waves), 32 rows per block.
// W tile arrives in LDS via TDM tensor_load_to_lds with hardware row padding
// (pad_interval = 64 DWORDs = one 256B row, pad_amount = 4 DWORDs = 16B)
// reproducing the 136-bf16 bank-staggered stride.
// ---------------------------------------------------------------------------
__global__ __launch_bounds__(64)
void hgcn_hyplinear(const float* __restrict__ in, const bf16* __restrict__ wb,
                    const float* __restrict__ w32, const float* __restrict__ b,
                    float* __restrict__ out, int n) {
    __shared__ bf16  lds_w[DD * WSTR];     // W row-major, bf16 (TDM destination)
    __shared__ bf16  lds_x[TM * XSTR];     // x tile, bf16
    __shared__ float lds_mx[TM * MXSTR];   // mx tile, f32
    __shared__ float lds_hb[DD];           // hyp_b = proj(expmap0(b))
    __shared__ float lds_b[DD];
    __shared__ float s_y2;                 // ||hyp_b||^2

    const int tid     = threadIdx.x;
    const int rowbase = blockIdx.x * TM;
    const int wave    = tid >> 5;
    const int lane    = tid & 31;

#if __has_builtin(__builtin_amdgcn_tensor_load_to_lds)
    if (wave == 0) {   // one TDM issue per block (EXEC ignored by TDM)
        unsigned lds_off = (unsigned)(uintptr_t)&lds_w[0];
        unsigned long long ga = (unsigned long long)(uintptr_t)wb;
        // D# group0: count=1 | lds_addr | global_addr | type=2
        v4u g0 = { 1u, lds_off, (unsigned)ga,
                   (unsigned)(ga >> 32) | 0x80000000u };
        // D# group1: data_size=1(2B), pad_enable, pad_interval=5(64dw),
        //            pad_amount=3(4dw); dims/tiles = 128; stride0 = 128
        v8i g1 = { (int)((1u << 16) | (1u << 20) | (5u << 22) | (3u << 25)),
                   (int)(128u << 16),        // tensor_dim0[15:0] @ bits 63:48
                   (int)(128u << 16),        // tensor_dim1[15:0] @ bits 95:80
                   (int)(128u << 16),        // tile_dim0        @ bits 127:112
                   128,                      // tile_dim1 (tile_dim2 = 0)
                   128,                      // tensor_dim0_stride[31:0]
                   0, 0 };
        v4i g2 = {0, 0, 0, 0};
        v4i g3 = {0, 0, 0, 0};
#if __clang_major__ >= 23
        v8i g4 = {0, 0, 0, 0, 0, 0, 0, 0};
        __builtin_amdgcn_tensor_load_to_lds(g0, g1, g2, g3, g4, 0);
#else
        __builtin_amdgcn_tensor_load_to_lds(g0, g1, g2, g3, 0);
#endif
    }
#else
    for (int idx = tid; idx < DD * DD; idx += 64) {   // fallback staging
        int r = idx >> 7, c = idx & 127;
        lds_w[r * WSTR + c] = (bf16)w32[idx];
    }
#endif

    // Stage x tile (fp32 -> bf16), zero-pad beyond n (overlaps the W DMA).
    for (int idx = tid; idx < TM * DD; idx += 64) {
        int r = idx >> 7, c = idx & 127;
        int gr = rowbase + r;
        float v = (gr < n) ? in[(size_t)gr * DD + c] : 0.0f;
        lds_x[r * XSTR + c] = (bf16)v;
    }
    if (tid < 64) { lds_b[tid] = b[tid]; lds_b[tid + 64] = b[tid + 64]; }

#if __has_builtin(__builtin_amdgcn_tensor_load_to_lds)
    if (wave == 0) __builtin_amdgcn_s_wait_tensorcnt(0);
#endif
    __syncthreads();

    if (tid == 0) {                         // hyperbolic bias (once per block)
        float s = 0.f;
        for (int k = 0; k < DD; ++k) s += lds_b[k] * lds_b[k];
        float nb = fmaxf(sqrtf(s), MIN_NORM);
        float tn = tanhf(nb);
        float f  = tn / nb;
        float pf = (tn > MAXNORM) ? MAXNORM / tn : 1.0f;
        float hn = fminf(tn, MAXNORM);
        s_y2 = hn * hn;
        for (int k = 0; k < DD; ++k) lds_hb[k] = lds_b[k] * f * pf;
    }

    // ---------------- GEMM: mx = x @ w.T via v_wmma_f32_16x16x32_bf16 -------
    {
        const int trow  = wave * 16;                  // 16-row tile per wave
        const int am    = trow + (lane & 15);         // A: M = lane&15 both halves
        const int akoff = (lane >= 16) ? 8 : 0;       // upper lanes: K+8 / K+24
        const int bn    = (lane & 15);                // B: N = lane&15
        const int bkoff = (lane >= 16) ? 8 : 0;       // upper lanes: K+8 / K+24

        v8f acc[8];
        #pragma unroll
        for (int ct = 0; ct < 8; ++ct) acc[ct] = (v8f){0,0,0,0,0,0,0,0};

        #pragma unroll
        for (int kt = 0; kt < 4; ++kt) {              // K = 128 = 4 x 32
            const int kb = kt * 32;
            v8bf alo = *(const v8bf*)&lds_x[am * XSTR + kb + akoff];
            v8bf ahi = *(const v8bf*)&lds_x[am * XSTR + kb + 16 + akoff];
            v16bf a = __builtin_shufflevector(alo, ahi,
                        0,1,2,3,4,5,6,7,8,9,10,11,12,13,14,15);
            #pragma unroll
            for (int ct = 0; ct < 8; ++ct) {          // 8 x 16 output columns
                const int wr = ct * 16 + bn;          // W row = output dim
                v8bf blo = *(const v8bf*)&lds_w[wr * WSTR + kb + bkoff];
                v8bf bhi = *(const v8bf*)&lds_w[wr * WSTR + kb + bkoff + 16];
                v16bf bb = __builtin_shufflevector(blo, bhi,
                            0,1,2,3,4,5,6,7,8,9,10,11,12,13,14,15);
                acc[ct] = __builtin_amdgcn_wmma_f32_16x16x32_bf16(
                              false, a, false, bb, (short)0, acc[ct], false, false);
            }
        }
        // D layout: VGPR r -> M=r (lanes 0-15) / M=8+r (lanes 16-31), N=lane&15
        const int rhalf = (lane >= 16) ? 8 : 0;
        #pragma unroll
        for (int ct = 0; ct < 8; ++ct)
            #pragma unroll
            for (int r = 0; r < 8; ++r)
                lds_mx[(trow + r + rhalf) * MXSTR + ct * 16 + (lane & 15)] = acc[ct][r];
    }
    __syncthreads();

    // ---------------- row-wise hyperbolic epilogue (1 thread per row) -------
    if (tid < TM) {
        int gr = rowbase + tid;
        if (gr < n) {
            float* mx = &lds_mx[tid * MXSTR];
            float mxn2 = 0.f;
            for (int k = 0; k < DD; ++k) mxn2 += mx[k] * mx[k];
            float mxn = fmaxf(sqrtf(mxn2), MIN_NORM);

            float xn2 = 0.f;                            // fp32 input norm
            const float* xr = &in[(size_t)gr * DD];
            for (int k = 0; k < DD; k += 4) {
                float4 v = *(const float4*)&xr[k];
                xn2 += v.x*v.x + v.y*v.y + v.z*v.z + v.w*v.w;
            }
            float xn  = fmaxf(sqrtf(xn2), MIN_NORM);
            float art = atanhf(fminf(xn, ART_CLAMP));    // sc = 1
            float tt  = tanhf(mxn / xn * art);
            float fac = (mxn2 == 0.f) ? 0.f : (tt / mxn);  // mobius_matvec scale
            float resn = fac * mxn;                        // == tt
            float pf  = (resn > MAXNORM) ? MAXNORM / resn : 1.0f;   // proj
            float s1  = fac * pf;
            float hn  = fminf(resn, MAXNORM);
            float x2  = hn * hn;

            float xy = 0.f;                               // h and <h, hyp_b>
            for (int k = 0; k < DD; ++k) {
                float h = mx[k] * s1; mx[k] = h; xy += h * lds_hb[k];
            }
            float y2  = s_y2;
            float ca  = 1.f + 2.f * xy + y2;              // c = 1
            float cb  = 1.f - x2;
            float den = fmaxf(1.f + 2.f * xy + x2 * y2, MIN_NORM);
            float inv = 1.f / den;
            float mn2 = 0.f;                              // mobius_add
            for (int k = 0; k < DD; ++k) {
                float m = (ca * mx[k] + cb * lds_hb[k]) * inv;
                mx[k] = m; mn2 += m * m;
            }
            float mn   = fmaxf(sqrtf(mn2), MIN_NORM);
            float pf2  = (mn > MAXNORM) ? MAXNORM / mn : 1.0f;   // proj
            float pn   = fmaxf(fminf(mn, MAXNORM), MIN_NORM);
            float lf   = atanhf(fminf(pn, ART_CLAMP)) / pn * pf2; // logmap0
            float* o = &out[(size_t)gr * DD];
            for (int k = 0; k < DD; k += 4) {
                float4 ov = {mx[k]*lf, mx[k+1]*lf, mx[k+2]*lf, mx[k+3]*lf};
                *(float4*)&o[k] = ov;
            }
        }
    }
}

// ---------------------------------------------------------------------------
// zero agg buffer
// ---------------------------------------------------------------------------
__global__ __launch_bounds__(256)
void hgcn_zero(float* __restrict__ p, int n) {
    size_t i = (size_t)blockIdx.x * 256 + threadIdx.x;   // one float4 each
    if (i < (size_t)n * 32) {
        float4 z = {0.f, 0.f, 0.f, 0.f};
        *(float4*)&p[i * 4] = z;
    }
}

// ---------------------------------------------------------------------------
// scatter: agg[dst] += xt[src] * w_e   — one wave per edge, lane owns 4 dims
// (xt and agg are ~51MB each -> L2-resident on MI455X's 192MB L2)
// ---------------------------------------------------------------------------
__global__ __launch_bounds__(256)
void hgcn_scatter(const float* __restrict__ xt, const int* __restrict__ eidx,
                  const float* __restrict__ ew, float* __restrict__ agg, int ne) {
    int wid  = (blockIdx.x * blockDim.x + threadIdx.x) >> 5;
    int lane = threadIdx.x & 31;
    if (wid >= ne) return;
    int   dst = eidx[wid];
    int   src = eidx[ne + wid];
    float wgt = ew[wid];
    const float4 v = *(const float4*)&xt[(size_t)src * DD + lane * 4];
    float* a = &agg[(size_t)dst * DD + lane * 4];
    atomicAdd(a + 0, v.x * wgt);
    atomicAdd(a + 1, v.y * wgt);
    atomicAdd(a + 2, v.z * wgt);
    atomicAdd(a + 3, v.w * wgt);
}

// ---------------------------------------------------------------------------
// hypagg+act: out = proj(expmap0(relu(logmap0(proj(expmap0(agg))))))
// ---------------------------------------------------------------------------
__global__ __launch_bounds__(256)
void hgcn_hypagg(const float* __restrict__ in, float* __restrict__ out, int n) {
    int wid  = (blockIdx.x * blockDim.x + threadIdx.x) >> 5;
    int lane = threadIdx.x & 31;
    if (wid >= n) return;
    const float4 v = *(const float4*)&in[(size_t)wid * DD + lane * 4];
    float s = v.x*v.x + v.y*v.y + v.z*v.z + v.w*v.w;
    #pragma unroll
    for (int off = 16; off; off >>= 1) s += __shfl_xor(s, off, 32);
    float nn  = fmaxf(sqrtf(s), MIN_NORM);
    float tn  = tanhf(nn);
    float sc0 = (tn / nn) * ((tn > MAXNORM) ? MAXNORM / tn : 1.0f);  // expmap0+proj
    float hnorm = fmaxf(fminf(tn, MAXNORM), MIN_NORM);
    float lf  = atanhf(fminf(hnorm, ART_CLAMP)) / hnorm;             // logmap0
    float a0 = fmaxf(v.x * sc0 * lf, 0.f);                           // relu
    float a1 = fmaxf(v.y * sc0 * lf, 0.f);
    float a2 = fmaxf(v.z * sc0 * lf, 0.f);
    float a3 = fmaxf(v.w * sc0 * lf, 0.f);
    float s2 = a0*a0 + a1*a1 + a2*a2 + a3*a3;
    #pragma unroll
    for (int off = 16; off; off >>= 1) s2 += __shfl_xor(s2, off, 32);
    float n2  = fmaxf(sqrtf(s2), MIN_NORM);
    float tn2 = tanhf(n2);
    float sc2 = (tn2 / n2) * ((tn2 > MAXNORM) ? MAXNORM / tn2 : 1.0f); // expmap0+proj
    float4 o = {a0*sc2, a1*sc2, a2*sc2, a3*sc2};
    *(float4*)&out[(size_t)wid * DD + lane * 4] = o;
}

// ---------------------------------------------------------------------------
extern "C" void kernel_launch(void* const* d_in, const int* in_sizes, int n_in,
                              void* d_out, int out_size, void* d_ws, size_t ws_size,
                              hipStream_t stream) {
    const float* x  = (const float*)d_in[0];
    const int*   ei = (const int*)  d_in[1];
    const float* ew = (const float*)d_in[2];
    const float* w1 = (const float*)d_in[3];
    const float* b1 = (const float*)d_in[4];
    const float* w2 = (const float*)d_in[5];
    const float* b2 = (const float*)d_in[6];
    float* out = (float*)d_out;

    const int n  = in_sizes[0] / DD;     // 100000
    const int ne = in_sizes[2];          // 1600000

    float* buf0 = (float*)d_ws;                    // N*D
    float* buf1 = buf0 + (size_t)n * DD;           // N*D
    bf16*  wbf  = (bf16*)(buf1 + (size_t)n * DD);  // 128x128 bf16 staging for TDM

    const int rowBlocks  = (n + 7) / 8;            // 8 waves/block, row per wave
    const int edgeBlocks = (ne + 7) / 8;
    const int zeroBlocks = (n * 32 + 255) / 256;
    const int linBlocks  = (n + TM - 1) / TM;
    const int wBlocks    = (DD * DD + 255) / 256;

    // ---- layer 0: encode ----
    hgcn_encode  <<<rowBlocks, 256, 0, stream>>>(x, buf0, n);
    // ---- layer 1 ----
    hgcn_w2bf16  <<<wBlocks,   256, 0, stream>>>(w1, wbf);
    hgcn_hyplinear<<<linBlocks, 64, 0, stream>>>(buf0, wbf, w1, b1, buf1, n); // xt1
    hgcn_zero    <<<zeroBlocks, 256, 0, stream>>>(buf0, n);                 // agg1 = 0
    hgcn_scatter <<<edgeBlocks, 256, 0, stream>>>(buf1, ei, ew, buf0, ne);  // agg1
    hgcn_hypagg  <<<rowBlocks, 256, 0, stream>>>(buf0, buf1, n);            // h1
    // ---- layer 2 ----
    hgcn_w2bf16  <<<wBlocks,   256, 0, stream>>>(w2, wbf);
    hgcn_hyplinear<<<linBlocks, 64, 0, stream>>>(buf1, wbf, w2, b2, buf0, n); // xt2
    hgcn_zero    <<<zeroBlocks, 256, 0, stream>>>(buf1, n);                 // agg2 = 0
    hgcn_scatter <<<edgeBlocks, 256, 0, stream>>>(buf0, ei, ew, buf1, ne);  // agg2
    hgcn_hypagg  <<<rowBlocks, 256, 0, stream>>>(buf1, out, n);             // output
}
